// DualDGCNNCA_37357625540995
// MI455X (gfx1250) — matrix operand order repaired and checked by reading, not compiled, not run on hardware
//
#include <hip/hip_runtime.h>
#include <hip/hip_bf16.h>

#define BN_EPS_ 1e-5f
#define SLOPE_  0.2f
#define KNN_    20
#define NPTS_   1024
#define BATCH_  8
#define HEADS_  8
#define NEG_INF_ (-3.4e38f)

typedef __attribute__((ext_vector_type(16))) _Float16 v16h;
typedef __attribute__((ext_vector_type(8)))  float    v8f;

__device__ __forceinline__ v8f wmma_fp16(v16h a, v16h b, v8f c) {
  // D = A(16x32,f16) * B(32x16,f16) + C(16x16,f32)
  return __builtin_amdgcn_wmma_f32_16x16x32_f16(
      /*neg_a=*/false, a, /*neg_b=*/false, b,
      /*c_mod=*/(short)0, c, /*reuse_a=*/false, /*reuse_b=*/false);
}

__device__ __forceinline__ float lrelu_(float x) { return x >= 0.f ? x : SLOPE_ * x; }

// Pack a 16-float contiguous run (base 16B-aligned) into a B/W fragment:
// element j holds K = off + j, run is contiguous.
__device__ __forceinline__ void pack16(v16h& f, const float* p) {
  float4 w0 = *(const float4*)(p + 0);
  float4 w1 = *(const float4*)(p + 4);
  float4 w2 = *(const float4*)(p + 8);
  float4 w3 = *(const float4*)(p + 12);
  f[0]=(_Float16)w0.x; f[1]=(_Float16)w0.y; f[2]=(_Float16)w0.z; f[3]=(_Float16)w0.w;
  f[4]=(_Float16)w1.x; f[5]=(_Float16)w1.y; f[6]=(_Float16)w1.z; f[7]=(_Float16)w1.w;
  f[8]=(_Float16)w2.x; f[9]=(_Float16)w2.y; f[10]=(_Float16)w2.z; f[11]=(_Float16)w2.w;
  f[12]=(_Float16)w3.x; f[13]=(_Float16)w3.y; f[14]=(_Float16)w3.z; f[15]=(_Float16)w3.w;
}

// Pack an A fragment from two contiguous 8-float runs at p and p+16
// (A element j: K = base + j for j<8, K = base + 16 + (j-8) for j>=8).
__device__ __forceinline__ void packA(v16h& f, const float* p) {
  float4 a0 = *(const float4*)(p + 0);
  float4 a1 = *(const float4*)(p + 4);
  float4 a2 = *(const float4*)(p + 16);
  float4 a3 = *(const float4*)(p + 20);
  f[0]=(_Float16)a0.x; f[1]=(_Float16)a0.y; f[2]=(_Float16)a0.z; f[3]=(_Float16)a0.w;
  f[4]=(_Float16)a1.x; f[5]=(_Float16)a1.y; f[6]=(_Float16)a1.z; f[7]=(_Float16)a1.w;
  f[8]=(_Float16)a2.x; f[9]=(_Float16)a2.y; f[10]=(_Float16)a2.z; f[11]=(_Float16)a2.w;
  f[12]=(_Float16)a3.x; f[13]=(_Float16)a3.y; f[14]=(_Float16)a3.z; f[15]=(_Float16)a3.w;
}

// ---------------------------------------------------------------------------
// kNN: per point n, top-20 of score = 2*<x_n,x_m> - ||x_m||^2  (same ordering
// as reference neg_d since -||x_n||^2 is a per-row constant).
// ---------------------------------------------------------------------------
template<int C>
__global__ __launch_bounds__(256) void knn_kernel(const float* __restrict__ x, long bstride,
                                                  int* __restrict__ idx) {
  const int TM = 128;
  __shared__ float tile[C][TM];
  __shared__ float sxx[TM];
  int tid = threadIdx.x;
  int n   = blockIdx.x * blockDim.x + tid;
  int b   = blockIdx.y;
  const float* xb = x + (long)b * bstride;

  float ctr[C];
#pragma unroll
  for (int c = 0; c < C; ++c) ctr[c] = xb[c * NPTS_ + n];

  float val[KNN_]; int ind[KNN_];
#pragma unroll
  for (int j = 0; j < KNN_; ++j) { val[j] = NEG_INF_; ind[j] = 0; }

  for (int m0 = 0; m0 < NPTS_; m0 += TM) {
    __syncthreads();
    for (int t = tid; t < C * TM; t += 256) {
      int c = t / TM, mm = t - c * TM;
      tile[c][mm] = xb[c * NPTS_ + m0 + mm];
    }
    __syncthreads();
    for (int mm = tid; mm < TM; mm += 256) {
      float s = 0.f;
#pragma unroll
      for (int c = 0; c < C; ++c) { float u = tile[c][mm]; s += u * u; }
      sxx[mm] = s;
    }
    __syncthreads();
    for (int mm = 0; mm < TM; ++mm) {
      float inner = 0.f;
#pragma unroll
      for (int c = 0; c < C; ++c) inner += ctr[c] * tile[c][mm];
      float score = 2.f * inner - sxx[mm];
      if (score > val[KNN_ - 1]) {
        float v = score; int id = m0 + mm;
#pragma unroll
        for (int j = 0; j < KNN_; ++j) {          // register bubble-insert
          bool gt = v > val[j];
          float tv = val[j]; int ti = ind[j];
          val[j] = gt ? v : tv;  ind[j] = gt ? id : ti;
          v      = gt ? tv : v;  id     = gt ? ti : id;
        }
      }
    }
  }
  int* op = idx + ((long)b * NPTS_ + n) * KNN_;
#pragma unroll
  for (int j = 0; j < KNN_; ++j) op[j] = ind[j];
}

// ---------------------------------------------------------------------------
// Fused edge-conv branch: edge = [x_j - x_n, x_n] (2C), y = edge @ w^T,
// BN + LeakyReLU, max over the 20 neighbors.  One wave handles
// (batch b, 8 points, 16 output channels): 160 rows = 10 WMMA M-tiles.
// All loads are unconditional (clamped address + cndmask for the C=3 tail).
// ---------------------------------------------------------------------------
template<int C>
__global__ __launch_bounds__(128) void edge_conv_kernel(
    const float* __restrict__ x, long xbstride,
    const int* __restrict__ idx,
    const float* __restrict__ w,          // (cout, 2C)
    const float* __restrict__ bng, const float* __restrict__ bnb,
    const float* __restrict__ bnm, const float* __restrict__ bnv,
    float* __restrict__ out, long obstride, int cout) {
  const int K2 = 2 * C;
  const int KS = (K2 + 31) / 32;
  const bool ALIGNED = (K2 % 32) == 0;     // compile-time per template
  const int NG = NPTS_ / 8;
  __shared__ int sidx[4][8 * KNN_];
  int lane  = threadIdx.x & 31;
  int wslot = threadIdx.x >> 5;
  int wid   = blockIdx.x * 4 + wslot;
  int otiles = cout >> 4;
  int ot = wid % otiles;
  int ng = (wid / otiles) % NG;
  int b  =  wid / (otiles * NG);

  const float* xb = x + (long)b * xbstride;
  const int* ib = idx + ((long)b * NPTS_ + ng * 8) * KNN_;
  int* s = sidx[wslot];
  for (int i = lane; i < 8 * KNN_; i += 32) s[i] = ib[i];

  int col = ot * 16 + (lane & 15);
  // B fragments (weights), cached across all 10 M-tiles
  v16h bf[KS];
#pragma unroll
  for (int ks = 0; ks < KS; ++ks) {
    if (ALIGNED) {
      pack16(bf[ks], w + (long)col * K2 + ks * 32 + (lane & 16));
    } else {
#pragma unroll
      for (int j = 0; j < 16; ++j) {
        int kk = ks * 32 + (lane & 16) + j;
        int kc = kk < K2 ? kk : 0;                 // clamp, load always safe
        float wv = w[(long)col * K2 + kc];
        bf[ks][j] = (_Float16)(kk < K2 ? wv : 0.f);
      }
    }
  }
  float scale = bng[col] * rsqrtf(bnv[col] + BN_EPS_);
  float shift = bnb[col] - bnm[col] * scale;

  float nmax[8];
#pragma unroll
  for (int i = 0; i < 8; ++i) nmax[i] = NEG_INF_;
  int half = (lane & 16) ? 8 : 0;

  for (int t = 0; t < 10; ++t) {
    int row   = t * 16 + (lane & 15);
    int nl    = row / KNN_;
    int nglob = ng * 8 + nl;
    int jn    = s[row];
    v8f acc = {0.f,0.f,0.f,0.f,0.f,0.f,0.f,0.f};
#pragma unroll
    for (int ks = 0; ks < KS; ++ks) {
      v16h af;
#pragma unroll
      for (int j = 0; j < 16; ++j) {
        int kk = ks * 32 + (lane & 16) + j;
        int kc = kk < C ? kk : kk - C;             // channel index in [0, C)
        if (!ALIGNED) kc = kk < K2 ? kc : 0;       // clamp tail (C==3 only)
        float ctrv = xb[kc * NPTS_ + nglob];       // unconditional loads
        float nbrv = xb[kc * NPTS_ + jn];
        float av = kk < C ? (nbrv - ctrv) : ctrv;  // cndmask, no branch
        if (!ALIGNED) av = kk < K2 ? av : 0.f;
        af[j] = (_Float16)av;
      }
      acc = wmma_fp16(af, bf[ks], acc);
    }
#pragma unroll
    for (int r = 0; r < 8; ++r) {
      int rr = t * 16 + r + half;
      int ni = rr / KNN_;
      float y = lrelu_(acc[r] * scale + shift);
#pragma unroll
      for (int q = 0; q < 8; ++q)                  // keep nmax in registers
        if (ni == q) nmax[q] = fmaxf(nmax[q], y);
    }
  }
#pragma unroll
  for (int ni = 0; ni < 8; ++ni) {
    float v = nmax[ni];
    v = fmaxf(v, __shfl_xor(v, 16));
    if (lane < 16)
      out[(long)b * obstride + (long)col * NPTS_ + ng * 8 + ni] = v;
  }
}

// ---------------------------------------------------------------------------
// Generic WMMA GEMM: O[row,col] = sum_k A[row,k] * W[col,k] (+bias)
//   AMODE 0: A row-major [M x K] (float4 runs); AMODE 1: channel-major
//            A[(l*K+k)*N + n] (b32 gathers, fully coalesced across lanes).
// K must be a multiple of 32 (true for every call site: 32/64/128/512).
//   omode 0: out row-major;  omode 1: channel-major out/resid.
// Optional BN, LeakyReLU, residual add.  One wave per 16x16 tile.
// ---------------------------------------------------------------------------
template<int AMODE>
__global__ __launch_bounds__(256) void gemm_wmma_kernel(
    const float* __restrict__ A,
    int M, int K, int Ncols, long a_bstride,
    const float* __restrict__ W,
    const float* __restrict__ bias,
    const float* __restrict__ bng, const float* __restrict__ bnb,
    const float* __restrict__ bnm, const float* __restrict__ bnv,
    int do_lrelu,
    const float* __restrict__ resid, long r_bstride,
    float* __restrict__ out, int omode, long o_bstride) {
  int lane = threadIdx.x & 31;
  int wid  = (blockIdx.x * blockDim.x + threadIdx.x) >> 5;
  int tn = Ncols >> 4;
  int tiles = (M >> 4) * tn;
  if (wid >= tiles) return;                 // whole-wave exit: EXEC stays all-ones
  int tm = wid / tn;
  int tc = wid - tm * tn;
  int row = tm * 16 + (lane & 15);
  int col = tc * 16 + (lane & 15);
  int l = row / NPTS_, n = row - (row / NPTS_) * NPTS_;

  const float* wbase = W + (long)col * K + ((lane & 16) >> 1);
  const float* abase0 = A + (long)row * K + ((lane & 16) >> 1);       // AMODE 0
  const float* abase1 = A + (long)l * a_bstride + n;                  // AMODE 1

  v8f acc = {0.f,0.f,0.f,0.f,0.f,0.f,0.f,0.f};
  for (int k0 = 0; k0 < K; k0 += 32) {
    v16h af, wf;
    if (AMODE == 0) {
      packA(af, abase0 + k0);
    } else {
#pragma unroll
      for (int j = 0; j < 16; ++j) {
        int ka = k0 + ((lane & 16) >> 1) + j + (j & 8);
        af[j] = (_Float16)abase1[(long)ka * NPTS_];   // coalesced across lanes
      }
    }
    packA(wf, wbase + k0);
    acc = wmma_fp16(af, wf, acc);
  }

  int half = (lane & 16) ? 8 : 0;
  float scale = 1.f, shift = 0.f;
  if (bng) { scale = bng[col] * rsqrtf(bnv[col] + BN_EPS_); shift = bnb[col] - bnm[col] * scale; }
  float bb = bias ? bias[col] : 0.f;
#pragma unroll
  for (int r = 0; r < 8; ++r) {
    int rrow = tm * 16 + r + half;
    float v = acc[r] + bb;
    if (bng) v = v * scale + shift;
    if (do_lrelu) v = lrelu_(v);
    int ll = rrow / NPTS_, nn = rrow - (rrow / NPTS_) * NPTS_;
    if (resid) v += resid[(long)ll * r_bstride + (long)col * NPTS_ + nn];
    if (omode) out[(long)ll * o_bstride + (long)col * NPTS_ + nn] = v;
    else       out[(long)rrow * Ncols + col] = v;
  }
}

// ---------------------------------------------------------------------------
// MHA core: seq length is the batch axis (L = M = 8).  One thread per (n, h).
// ---------------------------------------------------------------------------
template<int DH>
__global__ __launch_bounds__(256) void attn_kernel(
    const float* __restrict__ q, const float* __restrict__ k,
    const float* __restrict__ v, float* __restrict__ o) {
  int t = blockIdx.x * blockDim.x + threadIdx.x;
  if (t >= NPTS_ * HEADS_) return;
  int n = t % NPTS_, h = t / NPTS_;
  const int e = HEADS_ * DH;
  float rscale = rsqrtf((float)DH);
  float s[8][8];
#pragma unroll
  for (int l = 0; l < 8; ++l) {
    float ql[DH];
#pragma unroll
    for (int d = 0; d < DH; ++d) ql[d] = q[((long)l * NPTS_ + n) * e + h * DH + d];
#pragma unroll
    for (int m = 0; m < 8; ++m) {
      float a = 0.f;
#pragma unroll
      for (int d = 0; d < DH; ++d) a += ql[d] * k[((long)m * NPTS_ + n) * e + h * DH + d];
      s[l][m] = a * rscale;
    }
  }
#pragma unroll
  for (int l = 0; l < 8; ++l) {
    float mx = s[l][0];
#pragma unroll
    for (int m = 1; m < 8; ++m) mx = fmaxf(mx, s[l][m]);
    float sum = 0.f;
#pragma unroll
    for (int m = 0; m < 8; ++m) { float ev = __expf(s[l][m] - mx); s[l][m] = ev; sum += ev; }
    float inv = 1.f / sum;
#pragma unroll
    for (int m = 0; m < 8; ++m) s[l][m] *= inv;
  }
#pragma unroll
  for (int d = 0; d < DH; ++d) {
    float vv[8];
#pragma unroll
    for (int m = 0; m < 8; ++m) vv[m] = v[((long)m * NPTS_ + n) * e + h * DH + d];
#pragma unroll
    for (int l = 0; l < 8; ++l) {
      float a = 0.f;
#pragma unroll
      for (int m = 0; m < 8; ++m) a += s[l][m] * vv[m];
      o[((long)l * NPTS_ + n) * e + h * DH + d] = a;
    }
  }
}

// max + mean pooling over the point axis
__global__ __launch_bounds__(256) void pool_kernel(const float* __restrict__ y, int emb,
                                                   float* __restrict__ z) {
  int t = blockIdx.x * blockDim.x + threadIdx.x;
  if (t >= BATCH_ * emb) return;
  int b = t / emb, c = t - b * emb;
  const float* p = y + ((long)b * emb + c) * NPTS_;
  float mx = NEG_INF_, sm = 0.f;
  for (int n = 0; n < NPTS_; ++n) { float u = p[n]; mx = fmaxf(mx, u); sm += u; }
  z[(long)b * (2 * emb) + c]       = mx;
  z[(long)b * (2 * emb) + emb + c] = sm * (1.f / NPTS_);
}

// small FC: one thread per output element (M = 8 only)
__global__ __launch_bounds__(256) void fc_kernel(
    const float* __restrict__ A, int M, int K, int Nc,
    const float* __restrict__ W, const float* __restrict__ bias,
    const float* __restrict__ bng, const float* __restrict__ bnb,
    const float* __restrict__ bnm, const float* __restrict__ bnv,
    int do_lrelu, float* __restrict__ out) {
  int t = blockIdx.x * blockDim.x + threadIdx.x;
  if (t >= M * Nc) return;
  int r = t / Nc, c = t - r * Nc;
  float acc = 0.f;
  for (int k = 0; k < K; ++k) acc += A[(long)r * K + k] * W[(long)c * K + k];
  if (bias) acc += bias[c];
  if (bng) { float sc = bng[c] * rsqrtf(bnv[c] + BN_EPS_); acc = (acc - bnm[c]) * sc + bnb[c]; }
  if (do_lrelu) acc = lrelu_(acc);
  out[t] = acc;
}

// ---------------------------------------------------------------------------
extern "C" void kernel_launch(void* const* d_in, const int* in_sizes, int n_in,
                              void* d_out, int out_size, void* d_ws, size_t ws_size,
                              hipStream_t stream) {
  (void)in_sizes; (void)n_in; (void)out_size; (void)ws_size;
  const float* x0 = (const float*)d_in[0];
  auto P = [&](int i) { return (const float*)d_in[i]; };

  float* ws = (float*)d_ws;
  size_t off = 0;
  auto alloc = [&](size_t elems) { float* p = ws + off; off += (elems + 63) & ~(size_t)63; return p; };
  float* f    = alloc((size_t)BATCH_ * 512  * NPTS_);   // concat feature buffer
  float* ybuf = alloc((size_t)BATCH_ * 1024 * NPTS_);
  float* xl   = alloc((size_t)BATCH_ * 128  * NPTS_);
  float* xr   = alloc((size_t)BATCH_ * 128  * NPTS_);
  float* qb   = alloc((size_t)BATCH_ * NPTS_ * 128);
  float* kb   = alloc((size_t)BATCH_ * NPTS_ * 128);
  float* vb   = alloc((size_t)BATCH_ * NPTS_ * 128);
  float* ob   = alloc((size_t)BATCH_ * NPTS_ * 128);
  int*   idx  = (int*)alloc((size_t)BATCH_ * NPTS_ * KNN_);
  float* z    = alloc((size_t)BATCH_ * 2048);
  float* h1   = alloc((size_t)BATCH_ * 512);
  float* h2   = alloc((size_t)BATCH_ * 256);

  auto gemm = [&](const float* A, int amode, int M, int K, int Nc, long a_bs,
                  const float* W, const float* bias,
                  const float* g, const float* bbn, const float* m, const float* vv,
                  int lrl, const float* resid, long r_bs,
                  float* out, int omode, long o_bs) {
    int tiles  = (M / 16) * (Nc / 16);
    int blocks = (tiles * 32 + 255) / 256;
    if (amode == 0)
      gemm_wmma_kernel<0><<<blocks, 256, 0, stream>>>(A, M, K, Nc, a_bs, W, bias,
                                                      g, bbn, m, vv, lrl, resid, r_bs,
                                                      out, omode, o_bs);
    else
      gemm_wmma_kernel<1><<<blocks, 256, 0, stream>>>(A, M, K, Nc, a_bs, W, bias,
                                                      g, bbn, m, vv, lrl, resid, r_bs,
                                                      out, omode, o_bs);
  };

  struct Blk { int cin, cout, pb, chl, chr; };
  const Blk blks[4] = { {3, 32, 1, 0, 256}, {32, 32, 19, 32, 288},
                        {32, 64, 37, 64, 320}, {64, 128, 55, 128, 384} };
  const int M = BATCH_ * NPTS_;

  for (int bi = 0; bi < 4; ++bi) {
    const Blk& B = blks[bi];
    const float* xin; long xin_bs;
    if (bi == 0) { xin = x0; xin_bs = 3L * NPTS_; }
    else { xin = f + (long)blks[bi - 1].chl * NPTS_; xin_bs = 512L * NPTS_; }

    // 1) kNN
    dim3 kg(NPTS_ / 256, BATCH_);
    if (B.cin == 3)       knn_kernel<3> <<<kg, 256, 0, stream>>>(xin, xin_bs, idx);
    else if (B.cin == 32) knn_kernel<32><<<kg, 256, 0, stream>>>(xin, xin_bs, idx);
    else                  knn_kernel<64><<<kg, 256, 0, stream>>>(xin, xin_bs, idx);

    // 2) edge conv branches -> xl, xr   (channel-major (B, cout, N))
    int waves = BATCH_ * (NPTS_ / 8) * (B.cout / 16);
    int eblk  = waves / 4;
    long obs  = (long)B.cout * NPTS_;
    auto edge = [&](const float* w, const float* g, const float* bb,
                    const float* m, const float* vv, float* out) {
      if (B.cin == 3)
        edge_conv_kernel<3> <<<eblk, 128, 0, stream>>>(xin, xin_bs, idx, w, g, bb, m, vv, out, obs, B.cout);
      else if (B.cin == 32)
        edge_conv_kernel<32><<<eblk, 128, 0, stream>>>(xin, xin_bs, idx, w, g, bb, m, vv, out, obs, B.cout);
      else
        edge_conv_kernel<64><<<eblk, 128, 0, stream>>>(xin, xin_bs, idx, w, g, bb, m, vv, out, obs, B.cout);
    };
    edge(P(B.pb + 0), P(B.pb + 1),  P(B.pb + 2),  P(B.pb + 3),  P(B.pb + 4),  xl);
    edge(P(B.pb + 9), P(B.pb + 10), P(B.pb + 11), P(B.pb + 12), P(B.pb + 13), xr);

    // 3) two MHAs (seq axis = batch of 8)
    int e = B.cout;
    long ebs = (long)e * NPTS_;
    int ablk = (NPTS_ * HEADS_ + 255) / 256;
    auto mha = [&](const float* qsrc, const float* kvsrc, int pw, float* fout) {
      const float* inw = P(pw + 0); const float* inb = P(pw + 1);
      const float* ow  = P(pw + 2); const float* obp = P(pw + 3);
      gemm(qsrc,  1, M, e, e, ebs, inw,                   inb,       nullptr,nullptr,nullptr,nullptr, 0, nullptr, 0, qb, 0, 0);
      gemm(kvsrc, 1, M, e, e, ebs, inw + (size_t)e * e,   inb + e,   nullptr,nullptr,nullptr,nullptr, 0, nullptr, 0, kb, 0, 0);
      gemm(kvsrc, 1, M, e, e, ebs, inw + (size_t)2*e*e,   inb + 2*e, nullptr,nullptr,nullptr,nullptr, 0, nullptr, 0, vb, 0, 0);
      int dh = e / HEADS_;
      if (dh == 4)       attn_kernel<4> <<<ablk, 256, 0, stream>>>(qb, kb, vb, ob);
      else if (dh == 8)  attn_kernel<8> <<<ablk, 256, 0, stream>>>(qb, kb, vb, ob);
      else               attn_kernel<16><<<ablk, 256, 0, stream>>>(qb, kb, vb, ob);
      // out proj + residual (= xr for BOTH branches, faithful to source) -> f slice
      gemm(ob, 0, M, e, e, 0, ow, obp, nullptr,nullptr,nullptr,nullptr, 0,
           xr, ebs, fout, 1, 512L * NPTS_);
    };
    mha(xr, xl, B.pb + 5,  f + (long)B.chl * NPTS_);   // al: q=x2t, k/v=x1t
    mha(xl, xr, B.pb + 14, f + (long)B.chr * NPTS_);   // ar: q=x1t, k/v=x2t
  }

  // 4) final conv 512 -> 1024 with BN + lrelu, channel-major output
  gemm(f, 1, M, 512, 1024, 512L * NPTS_, P(73), nullptr,
       P(74), P(75), P(76), P(77), 1, nullptr, 0, ybuf, 1, 1024L * NPTS_);

  // 5) max + mean pool -> z (8, 2048)
  pool_kernel<<<(BATCH_ * 1024 + 255) / 256, 256, 0, stream>>>(ybuf, 1024, z);

  // 6) FC head
  fc_kernel<<<(BATCH_ * 512 + 255) / 256, 256, 0, stream>>>(
      z, BATCH_, 2048, 512, P(78), nullptr, P(79), P(80), P(81), P(82), 1, h1);
  fc_kernel<<<(BATCH_ * 256 + 255) / 256, 256, 0, stream>>>(
      h1, BATCH_, 512, 256, P(83), P(84), P(85), P(86), P(87), P(88), 1, h2);
  fc_kernel<<<(BATCH_ * 40 + 255) / 256, 256, 0, stream>>>(
      h2, BATCH_, 256, 40, P(89), P(90), nullptr, nullptr, nullptr, nullptr, 0,
      (float*)d_out);
}
